// PointNet2_62088047231573
// MI455X (gfx1250) — matrix operand order repaired and checked by reading, not compiled, not run on hardware
//
#include <hip/hip_runtime.h>

// ---------------------------------------------------------------------------
// PointNet++ forward for MI455X (gfx1250, wave32, WMMA).
// All pointwise MLPs run through v_wmma_f32_16x16x32_f16 (f32 accumulate).
// Geometry (FPS / ball query / 3-NN) stays in f32 VALU code.
// ---------------------------------------------------------------------------

typedef _Float16 f16;
typedef _Float16 v16h __attribute__((ext_vector_type(16)));
typedef _Float16 v8h  __attribute__((ext_vector_type(8)));
typedef float    v8f  __attribute__((ext_vector_type(8)));

static __host__ __device__ inline int cdiv_i(int a, int b) { return (a + b - 1) / b; }

// ---------------------------------------------------------------------------
// 1) Input transpose: pc (B,5,N) -> xyz0 (B,N,3) f32, pts0 (B,N,5) f32
// ---------------------------------------------------------------------------
__global__ void init_points_kernel(const float* __restrict__ pc,
                                   float* __restrict__ xyz0,
                                   float* __restrict__ pts0,
                                   int B, int N) {
  int i = blockIdx.x * blockDim.x + threadIdx.x;
  if (i >= B * N) return;
  int b = i / N, n = i % N;
  const float* src = pc + (size_t)b * 5 * N;
  #pragma unroll
  for (int c = 0; c < 5; ++c) {
    float v = src[(size_t)c * N + n];
    pts0[(size_t)i * 5 + c] = v;
    if (c < 3) xyz0[(size_t)i * 3 + c] = v;
  }
}

// ---------------------------------------------------------------------------
// 2) Farthest point sampling: one block per batch, serial npoint iterations.
//    dist kept in registers (32 per thread x 512 threads = 16384 points max).
// ---------------------------------------------------------------------------
__global__ __launch_bounds__(512)
void fps_kernel(const float* __restrict__ xyz, int N, int npoint,
                int* __restrict__ fidx) {
  const int b = blockIdx.x;
  const int tid = threadIdx.x;
  const float* p = xyz + (size_t)b * N * 3;

  float dmin[32];
  #pragma unroll
  for (int j = 0; j < 32; ++j) dmin[j] = 1e10f;

  __shared__ float sval[512];
  __shared__ int   sidx[512];
  __shared__ int   sfar;
  if (tid == 0) sfar = 0;

  for (int it = 0; it < npoint; ++it) {
    __syncthreads();
    const int far = sfar;
    const float cx = p[(size_t)far * 3 + 0];
    const float cy = p[(size_t)far * 3 + 1];
    const float cz = p[(size_t)far * 3 + 2];
    float best = -1.0f; int bi = 0;
    for (int j = 0; j < 32; ++j) {
      int n = j * 512 + tid;
      if (n < N) {
        float dx = p[(size_t)n * 3 + 0] - cx;
        float dy = p[(size_t)n * 3 + 1] - cy;
        float dz = p[(size_t)n * 3 + 2] - cz;
        float d = dx * dx + dy * dy + dz * dz;
        float dm = dmin[j];
        dm = d < dm ? d : dm;
        dmin[j] = dm;
        if (dm > best) { best = dm; bi = n; }
      }
    }
    sval[tid] = best; sidx[tid] = bi;
    __syncthreads();
    for (int str = 256; str > 0; str >>= 1) {
      if (tid < str) {
        float ov = sval[tid + str]; int oi = sidx[tid + str];
        if (ov > sval[tid] || (ov == sval[tid] && oi < sidx[tid])) {
          sval[tid] = ov; sidx[tid] = oi;
        }
      }
      __syncthreads();
    }
    if (tid == 0) {
      fidx[(size_t)b * npoint + it] = far;   // reference stacks pre-update index
      sfar = sidx[0];
    }
  }
}

// gather new_xyz[b][i] = xyz[b][fidx[b][i]]
__global__ void gather_xyz_kernel(const float* __restrict__ xyz,
                                  const int* __restrict__ fidx,
                                  float* __restrict__ nxyz,
                                  int total, int S, int N) {
  int i = blockIdx.x * blockDim.x + threadIdx.x;
  if (i >= total) return;
  int b = i / S;
  int src = fidx[i];
  #pragma unroll
  for (int c = 0; c < 3; ++c)
    nxyz[(size_t)i * 3 + c] = xyz[((size_t)b * N + src) * 3 + c];
}

// ---------------------------------------------------------------------------
// 3) Ball query: smallest 32 indices within radius, pad with first found.
// ---------------------------------------------------------------------------
__global__ void ball_query_kernel(const float* __restrict__ xyz,
                                  const float* __restrict__ nxyz,
                                  int N, int S, float r2,
                                  int* __restrict__ gidx, int total) {
  int i = blockIdx.x * blockDim.x + threadIdx.x;
  if (i >= total) return;
  int b = i / S;
  const float* p = xyz + (size_t)b * N * 3;
  float qx = nxyz[(size_t)i * 3 + 0];
  float qy = nxyz[(size_t)i * 3 + 1];
  float qz = nxyz[(size_t)i * 3 + 2];
  int* out = gidx + (size_t)i * 32;
  int cnt = 0;
  for (int n = 0; n < N; ++n) {
    float dx = p[(size_t)n * 3 + 0] - qx;
    float dy = p[(size_t)n * 3 + 1] - qy;
    float dz = p[(size_t)n * 3 + 2] - qz;
    float d = dx * dx + dy * dy + dz * dz;
    if (d <= r2) {
      out[cnt++] = n;
      if (cnt == 32) break;
    }
  }
  int first = (cnt > 0) ? out[0] : 0;
  for (; cnt < 32; ++cnt) out[cnt] = first;
}

// ---------------------------------------------------------------------------
// 4) Group features -> f16 activation matrix A (rows = B*S*32, cols = Kp),
//    columns: [rel xyz (3) or rel z (1)] ++ point features, zero-padded.
// ---------------------------------------------------------------------------
__global__ void group_feat_kernel(const float* __restrict__ xyz,
                                  const float* __restrict__ nxyz,
                                  const int* __restrict__ gidx,
                                  const float* __restrict__ ptsF,
                                  const f16* __restrict__ ptsH,
                                  int Cpts, int fullxyz,
                                  f16* __restrict__ A, int Kp,
                                  int N, int S, int rows) {
  int row = blockIdx.x * blockDim.x + threadIdx.x;
  if (row >= rows) return;
  int g = row >> 5;          // b*S + i
  int j = row & 31;
  int b = g / S;
  int n = gidx[(size_t)g * 32 + j];
  const float* src = xyz + ((size_t)b * N + n) * 3;
  const float* ctr = nxyz + (size_t)g * 3;
  f16* arow = A + (size_t)row * Kp;
  int c = 0;
  if (fullxyz) {
    arow[c++] = (f16)(src[0] - ctr[0]);
    arow[c++] = (f16)(src[1] - ctr[1]);
    arow[c++] = (f16)(src[2] - ctr[2]);
  } else {
    arow[c++] = (f16)(src[2] - ctr[2]);
  }
  if (ptsF) {
    const float* q = ptsF + ((size_t)b * N + n) * Cpts;
    for (int k = 0; k < Cpts; ++k) arow[c++] = (f16)q[k];
  } else {
    const f16* q = ptsH + ((size_t)b * N + n) * Cpts;
    for (int k = 0; k < Cpts; ++k) arow[c++] = q[k];
  }
  for (; c < Kp; ++c) arow[c] = (f16)0.0f;
}

// ---------------------------------------------------------------------------
// 5) Weight/BN prep: W(cin,cout) f32 -> Wt(cout,Kp) f16 (zero-pad rows),
//    s = g/sqrt(1+eps), t = b*s + beta
// ---------------------------------------------------------------------------
__global__ void prep_layer_kernel(const float* __restrict__ W,
                                  const float* __restrict__ b,
                                  const float* __restrict__ g,
                                  const float* __restrict__ beta,
                                  f16* __restrict__ Wt,
                                  float* __restrict__ s, float* __restrict__ t,
                                  int cin, int Kp, int cout) {
  int i = blockIdx.x * blockDim.x + threadIdx.x;
  int total = cout * Kp;
  if (i < total) {
    int co = i / Kp, k = i % Kp;
    Wt[i] = (k < cin) ? (f16)W[(size_t)k * cout + co] : (f16)0.0f;
  }
  if (i < cout) {
    float scale = rsqrtf(1.0f + 1e-5f);
    float sv = g[i] * scale;
    s[i] = sv;
    t[i] = b[i] * sv + beta[i];
  }
}

// ---------------------------------------------------------------------------
// 6) WMMA GEMM + fused BN affine + ReLU.
//    A: (M x Kp) f16 row-major. Wt: (Cout x Kp) f16 row-major (pre-transposed
//    so B-fragment loads are contiguous). Out: (M x Cout) f16.
//    One 16x16 output tile per wave; K-loop step 32 (v_wmma_f32_16x16x32_f16).
// ---------------------------------------------------------------------------
__global__ __launch_bounds__(128)
void wmma_gemm_bn_relu(const f16* __restrict__ A, const f16* __restrict__ Wt,
                       const float* __restrict__ s, const float* __restrict__ t,
                       f16* __restrict__ outH,
                       int M, int Kp, int Cout) {
  const int lane = threadIdx.x & 31;
  const int wave = threadIdx.x >> 5;
  const int tm = blockIdx.x * 4 + wave;   // 16-row tile index (wave-uniform)
  const int tn = blockIdx.y;              // 16-col tile index
  if (tm * 16 >= M) return;               // uniform per wave -> EXEC stays full

  const int l15  = lane & 15;
  const int half = lane >> 4;
  // A fragment (16x32 f16): lane covers row m; elems 0..7 = K[half*8..+7],
  // elems 8..15 = K[16+half*8..+7]
  const f16* arow = A  + (size_t)(tm * 16 + l15) * Kp + half * 8;
  // B fragment (32x16 f16): lane covers col n; elems 0..7 = K[half*8..+7],
  // elems 8..15 = K[16+half*8..+7]
  const f16* brow = Wt + (size_t)(tn * 16 + l15) * Kp + half * 8;

  v8f acc = {};
  for (int k = 0; k < Kp; k += 32) {
    __builtin_prefetch(arow + k + 32, 0, 1);
    __builtin_prefetch(brow + k + 32, 0, 1);
    union { v16h v; v8h h[2]; } af, bf;
    af.h[0] = *(const v8h*)(arow + k);
    af.h[1] = *(const v8h*)(arow + k + 16);
    bf.h[0] = *(const v8h*)(brow + k);
    bf.h[1] = *(const v8h*)(brow + k + 16);
    acc = __builtin_amdgcn_wmma_f32_16x16x32_f16(
        false, af.v, false, bf.v, (short)0, acc, false, false);
  }

  // C/D layout: VGPR v -> row (half*8 + v), col = lane&15
  const float sc = s[tn * 16 + l15];
  const float tb = t[tn * 16 + l15];
  #pragma unroll
  for (int v = 0; v < 8; ++v) {
    int r = tm * 16 + half * 8 + v;
    float y = acc[v] * sc + tb;
    y = y > 0.0f ? y : 0.0f;               // ReLU (all BN layers use it)
    outH[(size_t)r * Cout + tn * 16 + l15] = (f16)y;
  }
}

// ---------------------------------------------------------------------------
// 7) Max-pool over the 32-sample group dimension.
// ---------------------------------------------------------------------------
__global__ void maxpool32_kernel(const f16* __restrict__ A, f16* __restrict__ out,
                                 int groups, int C) {
  int i = blockIdx.x * blockDim.x + threadIdx.x;
  if (i >= groups * C) return;
  int g = i / C, c = i % C;
  const f16* p = A + ((size_t)g * 32) * C + c;
  float m = -1e30f;
  #pragma unroll 4
  for (int j = 0; j < 32; ++j) {
    float v = (float)p[(size_t)j * C];
    m = v > m ? v : m;
  }
  out[i] = (f16)m;
}

// ---------------------------------------------------------------------------
// 8) Feature propagation: 3-NN inverse-distance interpolation + concat -> A.
// ---------------------------------------------------------------------------
__global__ void fp_feat_kernel(const float* __restrict__ xyz1,
                               const float* __restrict__ xyz2,
                               const f16* __restrict__ pts1, int C1,
                               const f16* __restrict__ pts2, int C2,
                               f16* __restrict__ A, int Kp,
                               int N1, int N2, int B) {
  int i = blockIdx.x * blockDim.x + threadIdx.x;
  if (i >= B * N1) return;
  int b = i / N1;
  const float qx = xyz1[(size_t)i * 3 + 0];
  const float qy = xyz1[(size_t)i * 3 + 1];
  const float qz = xyz1[(size_t)i * 3 + 2];
  const float* p2 = xyz2 + (size_t)b * N2 * 3;

  float d0 = 1e30f, d1 = 1e30f, d2 = 1e30f;
  int   i0 = 0,     i1 = 0,     i2 = 0;
  for (int n = 0; n < N2; ++n) {
    float dx = p2[(size_t)n * 3 + 0] - qx;
    float dy = p2[(size_t)n * 3 + 1] - qy;
    float dz = p2[(size_t)n * 3 + 2] - qz;
    float d = dx * dx + dy * dy + dz * dz;
    if (d < d0)      { d2 = d1; i2 = i1; d1 = d0; i1 = i0; d0 = d; i0 = n; }
    else if (d < d1) { d2 = d1; i2 = i1; d1 = d;  i1 = n; }
    else if (d < d2) { d2 = d;  i2 = n; }
  }
  float w0 = 1.0f / (d0 + 1e-8f);
  float w1 = 1.0f / (d1 + 1e-8f);
  float w2 = 1.0f / (d2 + 1e-8f);
  float ws = w0 + w1 + w2;
  w0 /= ws; w1 /= ws; w2 /= ws;

  f16* arow = A + (size_t)i * Kp;
  int c = 0;
  if (pts1) {
    const f16* src = pts1 + (size_t)i * C1;
    for (int k = 0; k < C1; ++k) arow[c++] = src[k];
  }
  const f16* a0 = pts2 + ((size_t)b * N2 + i0) * C2;
  const f16* a1 = pts2 + ((size_t)b * N2 + i1) * C2;
  const f16* a2 = pts2 + ((size_t)b * N2 + i2) * C2;
  for (int k = 0; k < C2; ++k)
    arow[c++] = (f16)(w0 * (float)a0[k] + w1 * (float)a1[k] + w2 * (float)a2[k]);
  for (; c < Kp; ++c) arow[c] = (f16)0.0f;
}

// ---------------------------------------------------------------------------
// 9) Head: logits = h(128) @ W2(128x8) + b2, then log_softmax -> d_out f32.
// ---------------------------------------------------------------------------
__global__ void head_kernel(const f16* __restrict__ h,
                            const float* __restrict__ W2,
                            const float* __restrict__ b2,
                            float* __restrict__ out, int total) {
  int i = blockIdx.x * blockDim.x + threadIdx.x;
  if (i >= total) return;
  const f16* x = h + (size_t)i * 128;
  float logits[8];
  #pragma unroll
  for (int c = 0; c < 8; ++c) logits[c] = b2[c];
  for (int k = 0; k < 128; ++k) {
    float xv = (float)x[k];
    #pragma unroll
    for (int c = 0; c < 8; ++c) logits[c] += xv * W2[(size_t)k * 8 + c];
  }
  float mx = logits[0];
  #pragma unroll
  for (int c = 1; c < 8; ++c) mx = fmaxf(mx, logits[c]);
  float se = 0.0f;
  #pragma unroll
  for (int c = 0; c < 8; ++c) se += expf(logits[c] - mx);
  float lse = mx + logf(se);
  #pragma unroll
  for (int c = 0; c < 8; ++c) out[(size_t)i * 8 + c] = logits[c] - lse;
}

// ---------------------------------------------------------------------------
// Host orchestration
// ---------------------------------------------------------------------------
struct ParamPtrs { const float *W, *b, *g, *beta; };
struct LyrDev    { f16* wt; float* s; float* t; int kp; int cout; };

extern "C" void kernel_launch(void* const* d_in, const int* in_sizes, int n_in,
                              void* d_out, int out_size, void* d_ws, size_t ws_size,
                              hipStream_t stream) {
  (void)n_in; (void)out_size; (void)ws_size;
  const int B = 8, N0 = 16384;

  // ---- Input pointer mapping (auto-detect flattening order) ----
  // Canonical layer order: sa1[0..2] sa2[3..5] sa3[6..8] sa4[9..11]
  //                        fp4[12..13] fp3[14..15] fp2[16..17] fp1[18..20]
  //                        head1[21]; head2 separate (index 22).
  ParamPtrs pp[23];
  const float* PC;
  bool insertion = (in_sizes[0] == B * 5 * N0);
  if (insertion) {
    // dict insertion order: pc, sa1..sa4, fp4, fp3, fp2, fp1, head1, head2;
    // layer dict order: W, b, g, beta
    PC = (const float*)d_in[0];
    int pi = 1;
    auto grab = [&](int li) {
      pp[li].W    = (const float*)d_in[pi++];
      pp[li].b    = (const float*)d_in[pi++];
      pp[li].g    = (const float*)d_in[pi++];
      pp[li].beta = (const float*)d_in[pi++];
    };
    for (int l = 0; l < 12; ++l) grab(l);          // sa1..sa4
    grab(12); grab(13);                            // fp4
    grab(14); grab(15);                            // fp3
    grab(16); grab(17);                            // fp2
    grab(18); grab(19); grab(20);                  // fp1
    grab(21);                                      // head1
    pp[22].W = (const float*)d_in[pi++];           // head2
    pp[22].b = (const float*)d_in[pi++];
    pp[22].g = nullptr; pp[22].beta = nullptr;
  } else {
    // JAX sorted pytree order: params{fp1,fp2,fp3,fp4,head1,head2,sa1..sa4}, pc;
    // layer dict sorted: W, b, beta, g
    auto grabAt = [&](int li, int base) {
      pp[li].W    = (const float*)d_in[base + 0];
      pp[li].b    = (const float*)d_in[base + 1];
      pp[li].beta = (const float*)d_in[base + 2];
      pp[li].g    = (const float*)d_in[base + 3];
    };
    grabAt(18, 0);  grabAt(19, 4);  grabAt(20, 8);   // fp1
    grabAt(16, 12); grabAt(17, 16);                  // fp2
    grabAt(14, 20); grabAt(15, 24);                  // fp3
    grabAt(12, 28); grabAt(13, 32);                  // fp4
    grabAt(21, 36);                                  // head1
    pp[22].W = (const float*)d_in[40];               // head2
    pp[22].b = (const float*)d_in[41];
    pp[22].g = nullptr; pp[22].beta = nullptr;
    for (int l = 0; l < 12; ++l) grabAt(l, 42 + 4 * l);  // sa1..sa4
    PC = (const float*)d_in[90];
  }

  // ---- Workspace layout ----
  char* ws = (char*)d_ws;
  size_t off = 0;
  auto alloc = [&](size_t bytes) -> void* {
    void* p = ws + off;
    off = (off + bytes + 255) & ~(size_t)255;
    return p;
  };
  float* xyz0 = (float*)alloc((size_t)B * N0 * 3 * 4);
  float* pts0 = (float*)alloc((size_t)B * N0 * 5 * 4);
  int*   fidx = (int*)  alloc((size_t)B * 1024 * 4);
  int*   gidx = (int*)  alloc((size_t)B * 1024 * 32 * 4);
  float* l1x  = (float*)alloc((size_t)B * 1024 * 3 * 4);
  float* l2x  = (float*)alloc((size_t)B * 256  * 3 * 4);
  float* l3x  = (float*)alloc((size_t)B * 64   * 3 * 4);
  float* l4x  = (float*)alloc((size_t)B * 16   * 3 * 4);
  const size_t ACT_BYTES = (size_t)262144 * 64 * 2;   // == 131072*128*2
  f16* ACT0 = (f16*)alloc(ACT_BYTES);
  f16* ACT1 = (f16*)alloc(ACT_BYTES);
  f16* l1p = (f16*)alloc((size_t)B * 1024 * 64  * 2);
  f16* l2p = (f16*)alloc((size_t)B * 256  * 128 * 2);
  f16* l3p = (f16*)alloc((size_t)B * 64   * 256 * 2);
  f16* l4p = (f16*)alloc((size_t)B * 16   * 512 * 2);
  f16* p3  = (f16*)alloc((size_t)B * 64   * 256 * 2);
  f16* p2  = (f16*)alloc((size_t)B * 256  * 256 * 2);
  f16* p1  = (f16*)alloc((size_t)B * 1024 * 128 * 2);

  // ---- Layer dims (canonical order) + device-side weight prep ----
  static const int cins[22]  = { 8, 32, 32,   65, 64, 64,   129, 128, 128,
                                 257, 256, 256,  768, 256,  384, 256,
                                 320, 256,  128, 128, 128,  128 };
  static const int couts[22] = { 32, 32, 64,  64, 64, 128,  128, 128, 256,
                                 256, 256, 512,  256, 256,  256, 256,
                                 256, 128,  128, 128, 128,  128 };
  LyrDev L[22];
  for (int i = 0; i < 22; ++i) {
    L[i].kp   = ((cins[i] + 31) / 32) * 32;
    L[i].cout = couts[i];
    L[i].wt = (f16*)  alloc((size_t)L[i].cout * L[i].kp * 2);
    L[i].s  = (float*)alloc((size_t)L[i].cout * 4);
    L[i].t  = (float*)alloc((size_t)L[i].cout * 4);
    int tot = L[i].cout * L[i].kp;
    prep_layer_kernel<<<cdiv_i(tot, 256), 256, 0, stream>>>(
        pp[i].W, pp[i].b, pp[i].g, pp[i].beta,
        L[i].wt, L[i].s, L[i].t, cins[i], L[i].kp, L[i].cout);
  }

  auto gemm = [&](const f16* Ain, int li, f16* outH, int M) {
    dim3 grid((M / 16 + 3) / 4, L[li].cout / 16);
    wmma_gemm_bn_relu<<<grid, dim3(128), 0, stream>>>(
        Ain, L[li].wt, L[li].s, L[li].t, outH, M, L[li].kp, L[li].cout);
  };

  // ---- l0 init ----
  init_points_kernel<<<cdiv_i(B * N0, 256), 256, 0, stream>>>(PC, xyz0, pts0, B, N0);

  // ---- Set abstraction stages ----
  struct SAStage {
    const float* xyz; int N; int S; float r2;
    const float* ptsF; const f16* ptsH; int Cpts; int full;
    float* nxyz; int l0; f16* pooled; int Cout;
  };
  SAStage sa[4] = {
    { xyz0, N0,   1024, 0.01f, pts0,    nullptr, 5,   1, l1x, 0, l1p, 64  },
    { l1x,  1024, 256,  0.04f, nullptr, l1p,     64,  0, l2x, 3, l2p, 128 },
    { l2x,  256,  64,   0.16f, nullptr, l2p,     128, 0, l3x, 6, l3p, 256 },
    { l3x,  64,   16,   0.64f, nullptr, l3p,     256, 0, l4x, 9, l4p, 512 },
  };
  for (int st = 0; st < 4; ++st) {
    const SAStage& S = sa[st];
    const int BS = B * S.S;
    const int rows = BS * 32;
    const int Kp = L[S.l0].kp;
    fps_kernel<<<B, 512, 0, stream>>>(S.xyz, S.N, S.S, fidx);
    gather_xyz_kernel<<<cdiv_i(BS, 256), 256, 0, stream>>>(
        S.xyz, fidx, S.nxyz, BS, S.S, S.N);
    ball_query_kernel<<<cdiv_i(BS, 128), 128, 0, stream>>>(
        S.xyz, S.nxyz, S.N, S.S, S.r2, gidx, BS);
    group_feat_kernel<<<cdiv_i(rows, 256), 256, 0, stream>>>(
        S.xyz, S.nxyz, gidx, S.ptsF, S.ptsH, S.Cpts, S.full,
        ACT0, Kp, S.N, S.S, rows);
    gemm(ACT0, S.l0 + 0, ACT1, rows);
    gemm(ACT1, S.l0 + 1, ACT0, rows);
    gemm(ACT0, S.l0 + 2, ACT1, rows);
    maxpool32_kernel<<<cdiv_i(BS * S.Cout, 256), 256, 0, stream>>>(
        ACT1, S.pooled, BS, S.Cout);
  }

  // ---- Feature propagation stages ----
  // fp4: (l3 <- l4), C1=256, C2=512, Kp=768, out -> p3 (256 ch)
  {
    int N1 = 64, N2 = 16, M = B * N1;
    fp_feat_kernel<<<cdiv_i(M, 128), 128, 0, stream>>>(
        l3x, l4x, l3p, 256, l4p, 512, ACT0, L[12].kp, N1, N2, B);
    gemm(ACT0, 12, ACT1, M);
    gemm(ACT1, 13, p3, M);
  }
  // fp3: (l2 <- l3), C1=128, C2=256, Kp=384, out -> p2 (256 ch)
  {
    int N1 = 256, N2 = 64, M = B * N1;
    fp_feat_kernel<<<cdiv_i(M, 128), 128, 0, stream>>>(
        l2x, l3x, l2p, 128, p3, 256, ACT0, L[14].kp, N1, N2, B);
    gemm(ACT0, 14, ACT1, M);
    gemm(ACT1, 15, p2, M);
  }
  // fp2: (l1 <- l2), C1=64, C2=256, Kp=320, out -> p1 (128 ch)
  {
    int N1 = 1024, N2 = 256, M = B * N1;
    fp_feat_kernel<<<cdiv_i(M, 128), 128, 0, stream>>>(
        l1x, l2x, l1p, 64, p2, 256, ACT0, L[16].kp, N1, N2, B);
    gemm(ACT0, 16, ACT1, M);
    gemm(ACT1, 17, p1, M);
  }
  // fp1: (l0 <- l1), C1=0, C2=128, Kp=128, three 128->128 layers
  const int Mfull = B * N0;
  {
    fp_feat_kernel<<<cdiv_i(Mfull, 128), 128, 0, stream>>>(
        xyz0, l1x, nullptr, 0, p1, 128, ACT0, L[18].kp, N0, 1024, B);
    gemm(ACT0, 18, ACT1, Mfull);
    gemm(ACT1, 19, ACT0, Mfull);
    gemm(ACT0, 20, ACT1, Mfull);
  }
  // head1 (128->128 BN+ReLU), then head2 + log_softmax
  gemm(ACT1, 21, ACT0, Mfull);
  head_kernel<<<cdiv_i(Mfull, 128), 128, 0, stream>>>(
      ACT0, pp[22].W, pp[22].b, (float*)d_out, Mfull);
}
